// MPNNPyG_81475529605767
// MI455X (gfx1250) — matrix operand order repaired and checked by reading, not compiled, not run on hardware
//
#include <hip/hip_runtime.h>
#include <hip/hip_bf16.h>

#define N_NODES   50000
#define N_EDGES   800000
#define ATOM_DIM  32
#define EDGE_DIM  16
#define HIDDEN    64
#define NUM_STEPS 3
#define NUM_GRAPHS 256

typedef __attribute__((ext_vector_type(16))) __bf16       bf16x16;
typedef __attribute__((ext_vector_type(8)))  float        v8f;
typedef __attribute__((ext_vector_type(4)))  unsigned int u32x4;
typedef unsigned short ushort_t;

// ---- bf16 helpers: native hardware converts (v_cvt_pk_bf16_f32) -----------
__device__ __forceinline__ __bf16 f2bf(float x) { return (__bf16)x; }
__device__ __forceinline__ unsigned short f2us(float x) {
  __bf16 b = (__bf16)x;
  unsigned short s; __builtin_memcpy(&s, &b, 2); return s;
}
__device__ __forceinline__ __bf16 us2bf(unsigned short s) {
  __bf16 b; __builtin_memcpy(&b, &s, 2); return b;
}

// Load a 16x32 bf16 A-fragment (or 32x16 B-fragment) half-row:
// two 16-byte chunks at +off and +off+16 elements (layout per ISA 7.12.2).
__device__ __forceinline__ bf16x16 load_frag16(const ushort_t* p, int off) {
  union { bf16x16 v; u32x4 q[2]; } u;
  u.q[0] = *reinterpret_cast<const u32x4*>(p + off);
  u.q[1] = *reinterpret_cast<const u32x4*>(p + off + 16);
  return u.v;
}
// Build an A-fragment from an f32 row with on-the-fly bf16 conversion.
__device__ __forceinline__ bf16x16 cvt_frag_f32(const float* p, int off0) {
  bf16x16 v;
  #pragma unroll
  for (int j = 0; j < 8; ++j) {
    v[j]     = f2bf(p[off0 + j]);
    v[8 + j] = f2bf(p[16 + off0 + j]);
  }
  return v;
}

#define WMMA_BF16(A, B, C) \
  __builtin_amdgcn_wmma_f32_16x16x32_bf16(false, (A), false, (B), (short)0, (C), false, false)

__device__ __forceinline__ float sigmoidf_(float x) { return 1.0f / (1.0f + __expf(-x)); }

// Non-returning fp32 scatter-add -> global_atomic_add_f32 (STOREcnt path)
__device__ __forceinline__ void atomic_add_f32(float* p, float v) {
  (void)__hip_atomic_fetch_add(p, v, __ATOMIC_RELAXED, __HIP_MEMORY_SCOPE_AGENT);
}

// ---- weight prep: convert/transpose weights to N-major K-contiguous bf16 --
__global__ __launch_bounds__(256) void k_prep(
    const float* __restrict__ W_embed, const float* __restrict__ W_m1,
    const float* __restrict__ W_m2,   const float* __restrict__ W_ih,
    const float* __restrict__ W_hh,   const float* __restrict__ W_h1,
    ushort_t* __restrict__ W_embedt, ushort_t* __restrict__ W_m1t,
    ushort_t* __restrict__ W_m2t,    ushort_t* __restrict__ Wih_bf,
    ushort_t* __restrict__ Whh_bf,   ushort_t* __restrict__ W_h1t) {
  int i = blockIdx.x * blockDim.x + threadIdx.x;
  if (i < 64 * 32)  { int n = i / 32,  k = i % 32;  W_embedt[i] = f2us(W_embed[k * 64 + n]); }
  if (i < 64 * 160) { int n = i / 160, k = i % 160;
                      W_m1t[i] = (k < 144) ? f2us(W_m1[k * 64 + n]) : (ushort_t)0; }
  if (i < 64 * 64)  { int n = i / 64,  k = i % 64;  W_m2t[i] = f2us(W_m2[k * 64 + n]); }
  if (i < 192 * 64) { Wih_bf[i] = f2us(W_ih[i]); Whh_bf[i] = f2us(W_hh[i]); } // already N-major
  if (i < 64 * 128) { int n = i / 128, k = i % 128; W_h1t[i] = f2us(W_h1[k * 64 + n]); }
}

__global__ void k_zero(float* __restrict__ p, int n) {
  int i = blockIdx.x * blockDim.x + threadIdx.x;
  if (i < n) p[i] = 0.0f;
}

// ---- h = x @ W_embed + b : one 16-node tile per wave, 4 WMMAs/tile --------
__global__ __launch_bounds__(256) void k_embed(
    const float* __restrict__ x, const ushort_t* __restrict__ W_embedt,
    const float* __restrict__ b_embed,
    float* __restrict__ h, ushort_t* __restrict__ h_bf) {
  __shared__ __align__(16) ushort_t lw[64 * 32];
  int tid = threadIdx.x;
  for (int i = tid; i < 64 * 32; i += 256) lw[i] = W_embedt[i];
  __syncthreads();
  int wave = tid >> 5, lane = tid & 31;
  int nl = lane & 15;
  int off0 = (lane < 16) ? 0 : 8;
  int kb   = (lane < 16) ? 0 : 8;
  int mb   = (lane < 16) ? 0 : 8;
  float be[4];
  #pragma unroll
  for (int t = 0; t < 4; ++t) be[t] = b_embed[t * 16 + nl];
  for (int tile = blockIdx.x * 8 + wave; tile < N_NODES / 16; tile += gridDim.x * 8) {
    int row = tile * 16 + nl;
    bf16x16 a = cvt_frag_f32(x + row * ATOM_DIM, off0);
    #pragma unroll
    for (int t = 0; t < 4; ++t) {
      v8f c;
      #pragma unroll
      for (int r = 0; r < 8; ++r) c[r] = be[t];
      bf16x16 b = load_frag16(lw + (t * 16 + nl) * 32, kb);
      c = WMMA_BF16(a, b, c);
      #pragma unroll
      for (int r = 0; r < 8; ++r) {
        int node = tile * 16 + mb + r, col = t * 16 + nl;
        h[node * HIDDEN + col]    = c[r];
        h_bf[node * HIDDEN + col] = f2us(c[r]);
      }
    }
  }
}

// ---- message MLP + scatter-add: 16 edges per wave, 28 WMMAs/tile ----------
__global__ __launch_bounds__(256) void k_msg(
    const ushort_t* __restrict__ h_bf, const int* __restrict__ edge_index,
    const float* __restrict__ edge_attr,
    const ushort_t* __restrict__ W_m1t, const ushort_t* __restrict__ W_m2t,
    const float* __restrict__ b_m1, const float* __restrict__ b_m2,
    float* __restrict__ agg) {
  __shared__ __align__(16) ushort_t lw1[64 * 160];       // 20 KB
  __shared__ __align__(16) ushort_t lw2[64 * 64];        //  8 KB
  __shared__ __align__(16) ushort_t lt_all[8 * 16 * 64]; // 16 KB, per-wave tiles
  int tid = threadIdx.x;
  for (int i = tid; i < 64 * 160; i += 256) lw1[i] = W_m1t[i];
  for (int i = tid; i < 64 * 64;  i += 256) lw2[i] = W_m2t[i];
  __syncthreads();
  int wave = tid >> 5, lane = tid & 31;
  int nl = lane & 15;
  int off0 = (lane < 16) ? 0 : 8;
  int kb   = (lane < 16) ? 0 : 8;
  int mb   = (lane < 16) ? 0 : 8;
  ushort_t* lt = lt_all + wave * (16 * 64);
  const int* src_idx = edge_index;
  const int* dst_idx = edge_index + N_EDGES;
  float bm1[4], bm2[4];
  #pragma unroll
  for (int t = 0; t < 4; ++t) { bm1[t] = b_m1[t * 16 + nl]; bm2[t] = b_m2[t * 16 + nl]; }
  for (int tile = blockIdx.x * 8 + wave; tile < N_EDGES / 16; tile += gridDim.x * 8) {
    int em = tile * 16 + nl;
    if (em + 4096 < N_EDGES) __builtin_prefetch(&src_idx[em + 4096], 0, 0);
    int s = src_idx[em], d = dst_idx[em];
    // A-fragments: K-chunks 0-1 = h[src], 2-3 = h[dst], 4 = edge_attr|zeros
    bf16x16 a[5];
    const ushort_t* ps = h_bf + s * HIDDEN;
    const ushort_t* pd = h_bf + d * HIDDEN;
    a[0] = load_frag16(ps, off0);
    a[1] = load_frag16(ps + 32, off0);
    a[2] = load_frag16(pd, off0);
    a[3] = load_frag16(pd + 32, off0);
    {
      bf16x16 v;
      const float* pe = edge_attr + em * EDGE_DIM;
      #pragma unroll
      for (int j = 0; j < 8; ++j) { v[j] = f2bf(pe[off0 + j]); v[8 + j] = us2bf(0); }
      a[4] = v;
    }
    // scatter rows for this tile (invariant across N-tiles): agg row bases
    float* pr[8];
    #pragma unroll
    for (int r = 0; r < 8; ++r)
      pr[r] = agg + dst_idx[tile * 16 + mb + r] * HIDDEN + nl;
    // layer 1: [16x160(bf16)] x [160x64] + b, ReLU -> LDS (transpose C/D->A)
    #pragma unroll
    for (int t = 0; t < 4; ++t) {
      v8f c;
      #pragma unroll
      for (int r = 0; r < 8; ++r) c[r] = bm1[t];
      #pragma unroll
      for (int kc = 0; kc < 5; ++kc) {
        bf16x16 b = load_frag16(lw1 + (t * 16 + nl) * 160 + kc * 32, kb);
        c = WMMA_BF16(a[kc], b, c);
      }
      #pragma unroll
      for (int r = 0; r < 8; ++r)
        lt[(mb + r) * 64 + t * 16 + nl] = f2us(fmaxf(c[r], 0.0f));
    }
    // single-wave LDS RAW: DS ops are in-order per wave on CDNA5
    asm volatile("s_wait_dscnt 0" ::: "memory");
    bf16x16 a2[2];
    a2[0] = load_frag16(lt + nl * 64, off0);
    a2[1] = load_frag16(lt + nl * 64 + 32, off0);
    // layer 2: [16x64] x [64x64] + b, then scatter-add into agg[dst]
    #pragma unroll
    for (int t = 0; t < 4; ++t) {
      v8f c;
      #pragma unroll
      for (int r = 0; r < 8; ++r) c[r] = bm2[t];
      #pragma unroll
      for (int kc = 0; kc < 2; ++kc) {
        bf16x16 b = load_frag16(lw2 + (t * 16 + nl) * 64 + kc * 32, kb);
        c = WMMA_BF16(a2[kc], b, c);
      }
      #pragma unroll
      for (int r = 0; r < 8; ++r)
        atomic_add_f32(pr[r] + t * 16, c[r]);
    }
  }
}

// ---- GRU cell: gi = agg@W_ih^T, gh = h@W_hh^T, gate math, h update --------
__global__ __launch_bounds__(256) void k_gru(
    const float* __restrict__ agg, float* __restrict__ h,
    ushort_t* __restrict__ h_bf,
    const ushort_t* __restrict__ Wih_bf, const ushort_t* __restrict__ Whh_bf,
    const float* __restrict__ b_ih, const float* __restrict__ b_hh) {
  __shared__ __align__(16) ushort_t lwih[192 * 64];  // 24 KB
  __shared__ __align__(16) ushort_t lwhh[192 * 64];  // 24 KB
  int tid = threadIdx.x;
  for (int i = tid; i < 192 * 64; i += 256) { lwih[i] = Wih_bf[i]; lwhh[i] = Whh_bf[i]; }
  __syncthreads();
  int wave = tid >> 5, lane = tid & 31;
  int nl = lane & 15;
  int off0 = (lane < 16) ? 0 : 8;
  int kb   = (lane < 16) ? 0 : 8;
  int mb   = (lane < 16) ? 0 : 8;
  float bi[12], bh[12];                    // [gg*4 + t]
  #pragma unroll
  for (int gg = 0; gg < 3; ++gg)
    #pragma unroll
    for (int t = 0; t < 4; ++t) {
      bi[gg * 4 + t] = b_ih[gg * 64 + t * 16 + nl];
      bh[gg * 4 + t] = b_hh[gg * 64 + t * 16 + nl];
    }
  for (int tile = blockIdx.x * 8 + wave; tile < N_NODES / 16; tile += gridDim.x * 8) {
    int row = tile * 16 + nl;
    const float* pa = agg + row * HIDDEN;
    const float* ph = h   + row * HIDDEN;
    bf16x16 aA[2], aH[2];
    aA[0] = cvt_frag_f32(pa, off0); aA[1] = cvt_frag_f32(pa + 32, off0);
    aH[0] = cvt_frag_f32(ph, off0); aH[1] = cvt_frag_f32(ph + 32, off0);
    #pragma unroll
    for (int t = 0; t < 4; ++t) {
      v8f gi[3], gh[3];
      #pragma unroll
      for (int gg = 0; gg < 3; ++gg) {      // r, z, n gates
        int ng = gg * 64 + t * 16 + nl;
        #pragma unroll
        for (int r = 0; r < 8; ++r) { gi[gg][r] = bi[gg * 4 + t]; gh[gg][r] = bh[gg * 4 + t]; }
        #pragma unroll
        for (int kc = 0; kc < 2; ++kc) {
          bf16x16 b1 = load_frag16(lwih + ng * 64 + kc * 32, kb);
          gi[gg] = WMMA_BF16(aA[kc], b1, gi[gg]);
          bf16x16 b2 = load_frag16(lwhh + ng * 64 + kc * 32, kb);
          gh[gg] = WMMA_BF16(aH[kc], b2, gh[gg]);
        }
      }
      #pragma unroll
      for (int r = 0; r < 8; ++r) {
        int node = tile * 16 + mb + r, col = t * 16 + nl;
        float hold = h[node * HIDDEN + col];
        float rr = sigmoidf_(gi[0][r] + gh[0][r]);
        float zz = sigmoidf_(gi[1][r] + gh[1][r]);
        float nn = tanhf(gi[2][r] + rr * gh[2][r]);
        float hn = (1.0f - zz) * nn + zz * hold;
        h[node * HIDDEN + col]    = hn;
        h_bf[node * HIDDEN + col] = f2us(hn);
      }
    }
  }
}

// ---- graph pooling: segment-sum h by batch + counts -----------------------
__global__ void k_pool(const float* __restrict__ h, const int* __restrict__ batch,
                       float* __restrict__ g_sum, float* __restrict__ cnt) {
  int i = blockIdx.x * blockDim.x + threadIdx.x;
  if (i < N_NODES * HIDDEN) {
    int nidx = i / HIDDEN, c = i % HIDDEN;
    int b = batch[nidx];
    atomic_add_f32(&g_sum[b * HIDDEN + c], h[i]);
    if (c == 0) atomic_add_f32(&cnt[b], 1.0f);
  }
}

// ---- head: [sum|mean](128) -> 64 ReLU -> 1 : one wave per 16 graphs -------
__global__ __launch_bounds__(32) void k_head(
    const float* __restrict__ g_sum, const float* __restrict__ cnt,
    const ushort_t* __restrict__ W_h1t,
    const float* __restrict__ b_h1, const float* __restrict__ W_h2,
    const float* __restrict__ b_h2, float* __restrict__ out) {
  __shared__ __align__(16) float lh[16 * 64];
  int lane = threadIdx.x & 31;
  int nl = lane & 15;
  int off0 = (lane < 16) ? 0 : 8;
  int kb   = (lane < 16) ? 0 : 8;
  int mb   = (lane < 16) ? 0 : 8;
  int tile = blockIdx.x;
  int g = tile * 16 + nl;
  float rinv = 1.0f / fmaxf(cnt[g], 1.0f);
  bf16x16 a[4];                       // K = 128: chunks 0-1 = sum, 2-3 = mean
  #pragma unroll
  for (int cc = 0; cc < 4; ++cc) {
    bf16x16 v;
    #pragma unroll
    for (int j = 0; j < 8; ++j) {
      int k0 = 32 * cc + off0 + j, k1 = k0 + 16;
      float f0 = (k0 < 64) ? g_sum[g * 64 + k0] : g_sum[g * 64 + k0 - 64] * rinv;
      float f1 = (k1 < 64) ? g_sum[g * 64 + k1] : g_sum[g * 64 + k1 - 64] * rinv;
      v[j] = f2bf(f0); v[8 + j] = f2bf(f1);
    }
    a[cc] = v;
  }
  #pragma unroll
  for (int t = 0; t < 4; ++t) {
    float bias = b_h1[t * 16 + nl];
    v8f c;
    #pragma unroll
    for (int r = 0; r < 8; ++r) c[r] = bias;
    #pragma unroll
    for (int kc = 0; kc < 4; ++kc) {
      bf16x16 b = load_frag16(W_h1t + (t * 16 + nl) * 128 + kc * 32, kb);
      c = WMMA_BF16(a[kc], b, c);
    }
    #pragma unroll
    for (int r = 0; r < 8; ++r)
      lh[(mb + r) * 64 + t * 16 + nl] = fmaxf(c[r], 0.0f);
  }
  asm volatile("s_wait_dscnt 0" ::: "memory");
  if (lane < 16) {
    float s = b_h2[0];
    #pragma unroll 8
    for (int n2 = 0; n2 < 64; ++n2) s += lh[lane * 64 + n2] * W_h2[n2];
    out[tile * 16 + lane] = s;
  }
}

extern "C" void kernel_launch(void* const* d_in, const int* in_sizes, int n_in,
                              void* d_out, int out_size, void* d_ws, size_t ws_size,
                              hipStream_t stream) {
  (void)in_sizes; (void)n_in; (void)out_size; (void)ws_size;
  const float* x          = (const float*)d_in[0];
  const int*   edge_index = (const int*)d_in[1];
  const float* edge_attr  = (const float*)d_in[2];
  const int*   batch      = (const int*)d_in[3];
  const float* W_embed    = (const float*)d_in[4];
  const float* b_embed    = (const float*)d_in[5];
  const float* W_m1       = (const float*)d_in[6];
  const float* b_m1       = (const float*)d_in[7];
  const float* W_m2       = (const float*)d_in[8];
  const float* b_m2       = (const float*)d_in[9];
  const float* W_ih       = (const float*)d_in[10];
  const float* b_ih       = (const float*)d_in[11];
  const float* W_hh       = (const float*)d_in[12];
  const float* b_hh       = (const float*)d_in[13];
  const float* W_h1       = (const float*)d_in[14];
  const float* b_h1       = (const float*)d_in[15];
  const float* W_h2       = (const float*)d_in[16];
  const float* b_h2       = (const float*)d_in[17];
  float* out = (float*)d_out;

  char* ws = (char*)d_ws;
  size_t off = 0;
  auto take = [&](size_t bytes) {
    char* p = ws + off;
    off = (off + bytes + 255) & ~(size_t)255;
    return p;
  };
  float*    h       = (float*)take((size_t)N_NODES * HIDDEN * 4);
  ushort_t* h_bf    = (ushort_t*)take((size_t)N_NODES * HIDDEN * 2);
  float*    agg     = (float*)take((size_t)N_NODES * HIDDEN * 4);
  float*    g_sum   = (float*)take((size_t)NUM_GRAPHS * HIDDEN * 4);
  float*    cnt     = (float*)take((size_t)NUM_GRAPHS * 4);
  ushort_t* W_embedt= (ushort_t*)take(64 * 32 * 2);
  ushort_t* W_m1t   = (ushort_t*)take(64 * 160 * 2);
  ushort_t* W_m2t   = (ushort_t*)take(64 * 64 * 2);
  ushort_t* Wih_bf  = (ushort_t*)take(192 * 64 * 2);
  ushort_t* Whh_bf  = (ushort_t*)take(192 * 64 * 2);
  ushort_t* W_h1t   = (ushort_t*)take(64 * 128 * 2);

  k_prep<<<48, 256, 0, stream>>>(W_embed, W_m1, W_m2, W_ih, W_hh, W_h1,
                                 W_embedt, W_m1t, W_m2t, Wih_bf, Whh_bf, W_h1t);
  k_embed<<<(N_NODES / 16 + 7) / 8, 256, 0, stream>>>(x, W_embedt, b_embed, h, h_bf);
  for (int step = 0; step < NUM_STEPS; ++step) {
    k_zero<<<(N_NODES * HIDDEN + 255) / 256, 256, 0, stream>>>(agg, N_NODES * HIDDEN);
    k_msg<<<(N_EDGES / 16 + 7) / 8, 256, 0, stream>>>(h_bf, edge_index, edge_attr,
                                                      W_m1t, W_m2t, b_m1, b_m2, agg);
    k_gru<<<(N_NODES / 16 + 7) / 8, 256, 0, stream>>>(agg, h, h_bf,
                                                      Wih_bf, Whh_bf, b_ih, b_hh);
  }
  k_zero<<<(NUM_GRAPHS * HIDDEN + 255) / 256, 256, 0, stream>>>(g_sum, NUM_GRAPHS * HIDDEN);
  k_zero<<<1, 256, 0, stream>>>(cnt, NUM_GRAPHS);
  k_pool<<<(N_NODES * HIDDEN + 255) / 256, 256, 0, stream>>>(h, batch, g_sum, cnt);
  k_head<<<NUM_GRAPHS / 16, 32, 0, stream>>>(g_sum, cnt, W_h1t, b_h1, W_h2, b_h2, out);
}
